// En_Block_62225486184868
// MI455X (gfx1250) — matrix operand order repaired and checked by reading, not compiled, not run on hardware
//
#include <hip/hip_runtime.h>
#include <math.h>

// ---------------------------------------------------------------------------
// Fused Swin-style transformer block for MI455X (gfx1250, wave32, WMMA).
// B=16384, N=4 (window), C=512, H=8, HD=64, HID=2048.
// All GEMMs in f16 via v_wmma_f32_16x16x32_f16 (fp32 accumulate); everything
// between the two residual adds stays in LDS.
// ---------------------------------------------------------------------------

typedef _Float16 v16h __attribute__((ext_vector_type(16)));
typedef _Float16 v8h  __attribute__((ext_vector_type(8)));
typedef float    v8f  __attribute__((ext_vector_type(8)));

#define C_DIM 512
#define HID_DIM 2048
#define MT 32          // token rows per workgroup (= 8 windows of 4)
#define LDH 520        // f16 LDS row stride in halves (1040B = 16B aligned, bank-skewed)
#define LDX 516        // f32 LDS row stride in floats (2064B = 16B aligned)

// LDS byte offsets (total 133120 B dynamic LDS)
#define OFF_H   0              // h / h2  f16 [32][520]  (33280 B)
#define OFF_Q   33280          // q, later o, later u-chunk  f16 [32][520]
#define OFF_K   66560          // k  f16 [32][520]
#define OFF_V   99840          // v  f16 [32][520]
#define OFF_X1  66560          // x1 f32 [32][516] aliases k+v after attention
#define LDS_BYTES 133120

// --- A fragment: 16x32 f16, row-major source (ISA 7.12.2 16-bit A layout) ---
// lane 0-15: row m=lane, K 0-7 in v0-3, K 16-23 in v4-7
// lane 16-31: row m=lane-16, K 8-15 and 24-31
__device__ __forceinline__ v16h frag_a(const _Float16* base, int ld, int m0, int k0, int lane) {
  const _Float16* p = base + (size_t)(m0 + (lane & 15)) * ld + (k0 + ((lane >> 4) << 3));
  v8h lo = *(const v8h*)p;          // K = k0+ksel .. +7
  v8h hi = *(const v8h*)(p + 16);   // K = k0+16+ksel .. +7
  v16h f;
#pragma unroll
  for (int i = 0; i < 8; ++i) { f[i] = lo[i]; f[i + 8] = hi[i]; }
  return f;
}

// --- B fragment: 32x16 f16 (K x N), column n = weight row (h @ W.T) ---
// lane 0-15: col n=lane, K 0-7 and 16-23; lane 16-31: col n=lane-16, K 8-15 and 24-31
__device__ __forceinline__ v16h frag_b(const _Float16* base, int ld, int n0, int k0, int lane) {
  const _Float16* p = base + (size_t)(n0 + (lane & 15)) * ld + (k0 + ((lane >> 4) << 3));
  v8h lo = *(const v8h*)p;
  v8h hi = *(const v8h*)(p + 16);
  v16h f;
#pragma unroll
  for (int i = 0; i < 8; ++i) { f[i] = lo[i]; f[i + 8] = hi[i]; }
  return f;
}

__device__ __forceinline__ v8f wmma16(v16h a, v16h b, v8f c) {
  // D = A x B + C, fp32 accumulate
  return __builtin_amdgcn_wmma_f32_16x16x32_f16(false, a, false, b, (short)0, c, false, false);
}

// ---------------------------------------------------------------------------
// Prep: convert all four weight matrices fp32 -> f16 into workspace.
// layout (halves): [0, 786432) qkv_w | 262144 proj_w | 1048576 fc1_w | 1048576 fc2_w
// ---------------------------------------------------------------------------
__global__ void cvt_w_kernel(const float* __restrict__ qkv_w, const float* __restrict__ proj_w,
                             const float* __restrict__ fc1_w, const float* __restrict__ fc2_w,
                             _Float16* __restrict__ w16) {
  size_t i = (size_t)blockIdx.x * blockDim.x + threadIdx.x;
  const size_t NQ = 1536u * 512u, NP = 512u * 512u, NF1 = 2048u * 512u, NT = NQ + NP + NF1 + 512u * 2048u;
  if (i >= NT) return;
  float v;
  if (i < NQ)                 v = qkv_w[i];
  else if (i < NQ + NP)       v = proj_w[i - NQ];
  else if (i < NQ + NP + NF1) v = fc1_w[i - NQ - NP];
  else                        v = fc2_w[i - NQ - NP - NF1];
  w16[i] = (_Float16)v;
}

// ---------------------------------------------------------------------------
// Fused block kernel. 256 threads = 8 waves, 32 token rows per workgroup.
// ---------------------------------------------------------------------------
__global__ __launch_bounds__(256) void swin_block_kernel(
    const float* __restrict__ x, const float* __restrict__ ln_g, const float* __restrict__ ln_b,
    const _Float16* __restrict__ wqkv, const _Float16* __restrict__ wproj,
    const float* __restrict__ proj_b, const float* __restrict__ rpb,
    const _Float16* __restrict__ wfc1, const float* __restrict__ fc1_b,
    const _Float16* __restrict__ wfc2, const float* __restrict__ fc2_b,
    float* __restrict__ out) {
  extern __shared__ char smem[];
  _Float16* sH  = (_Float16*)(smem + OFF_H);
  _Float16* sQ  = (_Float16*)(smem + OFF_Q);
  _Float16* sK  = (_Float16*)(smem + OFF_K);
  _Float16* sV  = (_Float16*)(smem + OFF_V);
  float*    sX1 = (float*)   (smem + OFF_X1);
  _Float16* sU  = sQ;

  const int tid = threadIdx.x;
  const int lane = tid & 31;
  const int w = tid >> 5;                 // wave id 0..7
  const int rowBase = blockIdx.x * MT;    // global token row of this tile

  // ---------------- Phase 0: LN1(x) -> sH (f16) ----------------
  for (int r = w; r < MT; r += 8) {
    const float* xr = x + (size_t)(rowBase + r) * C_DIM;
    float vals[16];
    float s = 0.f;
#pragma unroll
    for (int e = 0; e < 16; ++e) { vals[e] = xr[lane + 32 * e]; s += vals[e]; }
#pragma unroll
    for (int m = 16; m >= 1; m >>= 1) s += __shfl_xor(s, m, 32);
    float mean = s * (1.0f / 512.0f);
    float vv = 0.f;
#pragma unroll
    for (int e = 0; e < 16; ++e) { float d = vals[e] - mean; vv += d * d; }
#pragma unroll
    for (int m = 16; m >= 1; m >>= 1) vv += __shfl_xor(vv, m, 32);
    float rstd = rsqrtf(vv * (1.0f / 512.0f) + 1e-5f);
#pragma unroll
    for (int e = 0; e < 16; ++e) {
      int col = lane + 32 * e;
      sH[r * LDH + col] = (_Float16)((vals[e] - mean) * rstd * ln_g[col] + ln_b[col]);
    }
  }
  __syncthreads();

  // ---------------- Phase 1: qkv = h @ Wqkv^T  (WMMA) ----------------
  // output [32 x 1536] -> 2 x 96 = 192 tiles, 24 per wave
  for (int t = w; t < 192; t += 8) {
    int rt = t & 1, ct = t >> 1;
    v8f acc = {};
    for (int k0 = 0; k0 < C_DIM; k0 += 32) {
      __builtin_prefetch(wqkv + (size_t)(ct * 16 + (lane & 15)) * C_DIM + k0 + 128, 0, 0);
      v16h a = frag_a(sH, LDH, rt * 16, k0, lane);
      v16h b = frag_b(wqkv, C_DIM, ct * 16, k0, lane);
      acc = wmma16(a, b, acc);
    }
    int col = ct * 16 + (lane & 15);                        // 0..1535: [q|k|v]
    _Float16* dst = (col < 512) ? sQ : (col < 1024) ? sK : sV;
    int cc = col & 511;
    int r0 = rt * 16 + ((lane >> 4) << 3);
#pragma unroll
    for (int r = 0; r < 8; ++r) dst[(r0 + r) * LDH + cc] = (_Float16)acc[r];
  }
  __syncthreads();

  // ---------------- Phase 2: windowed attention (N=4, HD=64) ----------------
  // 8 windows x 8 heads x 4 query rows = 256 tasks, one per thread.
  {
    int qi = tid & 3, head = (tid >> 2) & 7, win = tid >> 5;
    int ri = win * 4 + qi;
    _Float16* qp = sQ + ri * LDH + head * 64;
    float sc[4];
#pragma unroll
    for (int j = 0; j < 4; ++j) {
      const _Float16* kp = sK + (win * 4 + j) * LDH + head * 64;
      float d = 0.f;
#pragma unroll 8
      for (int e = 0; e < 64; ++e) d += (float)qp[e] * (float)kp[e];
      // REL_IDX for WIN=2: y=n>>1, x=n&1; idx=(dy+1)*3+(dx+1)
      int idx = ((qi >> 1) - (j >> 1) + 1) * 3 + ((qi & 1) - (j & 1) + 1);
      sc[j] = d * 0.125f + rpb[idx * 8 + head];
    }
    float mx = fmaxf(fmaxf(sc[0], sc[1]), fmaxf(sc[2], sc[3]));
    float p0 = expf(sc[0] - mx), p1 = expf(sc[1] - mx), p2 = expf(sc[2] - mx), p3 = expf(sc[3] - mx);
    float inv = 1.0f / (p0 + p1 + p2 + p3);
    p0 *= inv; p1 *= inv; p2 *= inv; p3 *= inv;
    const _Float16* v0 = sV + (win * 4 + 0) * LDH + head * 64;
    const _Float16* v1 = sV + (win * 4 + 1) * LDH + head * 64;
    const _Float16* v2 = sV + (win * 4 + 2) * LDH + head * 64;
    const _Float16* v3 = sV + (win * 4 + 3) * LDH + head * 64;
#pragma unroll 8
    for (int e = 0; e < 64; ++e) {   // overwrite q with o (exclusive per task)
      float o = p0 * (float)v0[e] + p1 * (float)v1[e] + p2 * (float)v2[e] + p3 * (float)v3[e];
      qp[e] = (_Float16)o;
    }
  }
  __syncthreads();

  // ---------------- Phase 3: x1 = x + o @ Wproj^T + proj_b  (WMMA) ----------------
  // 2 x 32 = 64 tiles, 8 per wave; writes fp32 x1 into dead k/v region
  for (int t = w; t < 64; t += 8) {
    int rt = t & 1, ct = t >> 1;
    v8f acc = {};
    for (int k0 = 0; k0 < C_DIM; k0 += 32) {
      v16h a = frag_a(sQ, LDH, rt * 16, k0, lane);   // o lives in sQ
      v16h b = frag_b(wproj, C_DIM, ct * 16, k0, lane);
      acc = wmma16(a, b, acc);
    }
    int col = ct * 16 + (lane & 15);
    float pb = proj_b[col];
    int r0 = rt * 16 + ((lane >> 4) << 3);
#pragma unroll
    for (int r = 0; r < 8; ++r) {
      int row = r0 + r;
      sX1[row * LDX + col] = acc[r] + pb + x[(size_t)(rowBase + row) * C_DIM + col];
    }
  }
  __syncthreads();

  // ---------------- Phase 4: LN2(x1) -> sH (f16) ----------------
  for (int r = w; r < MT; r += 8) {
    float vals[16];
    float s = 0.f;
#pragma unroll
    for (int e = 0; e < 16; ++e) { vals[e] = sX1[r * LDX + lane + 32 * e]; s += vals[e]; }
#pragma unroll
    for (int m = 16; m >= 1; m >>= 1) s += __shfl_xor(s, m, 32);
    float mean = s * (1.0f / 512.0f);
    float vv = 0.f;
#pragma unroll
    for (int e = 0; e < 16; ++e) { float d = vals[e] - mean; vv += d * d; }
#pragma unroll
    for (int m = 16; m >= 1; m >>= 1) vv += __shfl_xor(vv, m, 32);
    float rstd = rsqrtf(vv * (1.0f / 512.0f) + 1e-5f);
#pragma unroll
    for (int e = 0; e < 16; ++e) {
      int col = lane + 32 * e;
      sH[r * LDH + col] = (_Float16)((vals[e] - mean) * rstd * ln_g[col] + ln_b[col]);
    }
  }
  __syncthreads();

  // ---------------- Phase 5/6: MLP, HID chunked by 512 ----------------
  v8f oacc[8];
#pragma unroll
  for (int i = 0; i < 8; ++i) oacc[i] = (v8f){};
  for (int c0 = 0; c0 < HID_DIM; c0 += 512) {
    // fc1 chunk: u = gelu(h2 @ Wfc1[c0:c0+512]^T + b1) -> sU (f16)
    for (int i = 0; i < 8; ++i) {
      int t = w * 8 + i;
      int rt = t & 1, ct = t >> 1;
      v8f acc = {};
      for (int k0 = 0; k0 < C_DIM; k0 += 32) {
        __builtin_prefetch(wfc1 + (size_t)(c0 + ct * 16 + (lane & 15)) * C_DIM + k0 + 128, 0, 0);
        v16h a = frag_a(sH, LDH, rt * 16, k0, lane);
        v16h b = frag_b(wfc1, C_DIM, c0 + ct * 16, k0, lane);
        acc = wmma16(a, b, acc);
      }
      int col = ct * 16 + (lane & 15);
      float b1 = fc1_b[c0 + col];
      int r0 = rt * 16 + ((lane >> 4) << 3);
#pragma unroll
      for (int r = 0; r < 8; ++r) {
        float u = acc[r] + b1;
        u = 0.5f * u * (1.0f + erff(u * 0.70710678118f));   // exact GELU
        sU[(r0 + r) * LDH + col] = (_Float16)u;
      }
    }
    __syncthreads();
    // fc2 partial: oacc += u @ Wfc2[:, c0:c0+512]^T
    for (int i = 0; i < 8; ++i) {
      int t = w * 8 + i;
      int rt = t & 1, ct = t >> 1;
      v8f acc = oacc[i];
      for (int k0 = 0; k0 < 512; k0 += 32) {
        v16h a = frag_a(sU, LDH, rt * 16, k0, lane);
        v16h b = frag_b(wfc2, HID_DIM, ct * 16, c0 + k0, lane);
        acc = wmma16(a, b, acc);
      }
      oacc[i] = acc;
    }
    __syncthreads();
  }

  // ---------------- Phase 7: out = x1 + fc2_out + fc2_b ----------------
  for (int i = 0; i < 8; ++i) {
    int t = w * 8 + i;
    int rt = t & 1, ct = t >> 1;
    int col = ct * 16 + (lane & 15);
    float b2 = fc2_b[col];
    int r0 = rt * 16 + ((lane >> 4) << 3);
#pragma unroll
    for (int r = 0; r < 8; ++r) {
      int row = r0 + r;
      out[(size_t)(rowBase + row) * C_DIM + col] = oacc[i][r] + b2 + sX1[row * LDX + col];
    }
  }
}

// ---------------------------------------------------------------------------
extern "C" void kernel_launch(void* const* d_in, const int* in_sizes, int n_in,
                              void* d_out, int out_size, void* d_ws, size_t ws_size,
                              hipStream_t stream) {
  const float* x      = (const float*)d_in[0];
  const float* ln_g   = (const float*)d_in[1];
  const float* ln_b   = (const float*)d_in[2];
  const float* qkv_w  = (const float*)d_in[3];
  const float* proj_w = (const float*)d_in[4];
  const float* proj_b = (const float*)d_in[5];
  const float* rpb    = (const float*)d_in[6];
  const float* fc1_w  = (const float*)d_in[7];
  const float* fc1_b  = (const float*)d_in[8];
  const float* fc2_w  = (const float*)d_in[9];
  const float* fc2_b  = (const float*)d_in[10];
  float* out = (float*)d_out;

  _Float16* w16 = (_Float16*)d_ws;
  const size_t NQ = 1536u * 512u, NP = 512u * 512u, NF1 = 2048u * 512u;
  _Float16* wqkv16 = w16;
  _Float16* wproj16 = w16 + NQ;
  _Float16* wfc1_16 = w16 + NQ + NP;
  _Float16* wfc2_16 = w16 + NQ + NP + NF1;

  // convert weights to f16 in workspace (~6.3 MB)
  {
    size_t total = NQ + NP + NF1 + 512u * 2048u;  // 3,145,728
    int blocks = (int)((total + 255) / 256);
    cvt_w_kernel<<<blocks, 256, 0, stream>>>(qkv_w, proj_w, fc1_w, fc2_w, w16);
  }

  // fused block: 65536 token rows / 32 per workgroup
  {
    dim3 grid(65536 / MT);
    swin_block_kernel<<<grid, 256, LDS_BYTES, stream>>>(
        x, ln_g, ln_b, wqkv16, wproj16, proj_b, rpb, wfc1_16, fc1_b, wfc2_16, fc2_b, out);
  }
}
